// Transformer_20023137534485
// MI455X (gfx1250) — matrix operand order repaired and checked by reading, not compiled, not run on hardware
//
#include <hip/hip_runtime.h>
#include <hip/hip_bf16.h>

#define BDIM   2
#define SEQ    2048
#define DMODEL 256
#define NH     8
#define DKV    32
#define DFFN   512
#define NLAYER 2
#define MROWS  (BDIM * SEQ)          // 4096
#define MASKV  (-1e-30f)
#define LN_EPS 1e-14f

typedef __attribute__((ext_vector_type(16))) __bf16 bf16x16;
typedef __attribute__((ext_vector_type(8)))  float  f32x8;

__device__ __forceinline__ unsigned short f2bf(float f) {
  unsigned int u = __float_as_uint(f);
  unsigned int r = (u + 0x7FFFu + ((u >> 16) & 1u)) >> 16;  // RNE
  return (unsigned short)r;
}

union AFrag { bf16x16 v; unsigned int u[8]; };

// A-fragment (16x32 bf16, MxK): lane = row (lane&15); K layout per ISA:
// lanes 0-15: V0..3 hold K=0..7, V4..7 hold K=16..23; lanes 16-31: +8.
__device__ __forceinline__ bf16x16 load_a_frag(const unsigned short* base, int ld) {
  const int lane = threadIdx.x & 31;
  const int row  = lane & 15;
  const int pk   = (lane >> 4) << 2;            // pair offset 0 or 4 (K 0/8)
  const unsigned int* p = (const unsigned int*)(base + row * ld);
  AFrag f;
#pragma unroll
  for (int v = 0; v < 4; ++v) f.u[v] = p[pk + v];
#pragma unroll
  for (int v = 0; v < 4; ++v) f.u[4 + v] = p[8 + pk + v];
  return f.v;
}

// B-fragment (32x16 bf16, KxN) loaded from B^T storage (N x K row-major):
// lane = column n; lanes 0-15 hold K=0..7,16..23, lanes 16-31 hold K=8..15,24..31.
__device__ __forceinline__ bf16x16 load_b_frag(const unsigned short* baseT, int ld) {
  const int lane = threadIdx.x & 31;
  const int n  = lane & 15;
  const int p0 = (lane >> 4) << 2;              // pair offset 0 or 4 (K 0/8)
  const unsigned int* p = (const unsigned int*)(baseT + n * ld);
  AFrag f;
#pragma unroll
  for (int v = 0; v < 4; ++v) f.u[v] = p[p0 + v];
#pragma unroll
  for (int v = 0; v < 4; ++v) f.u[4 + v] = p[8 + p0 + v];
  return f.v;
}

__device__ __forceinline__ f32x8 wmma_bf16(bf16x16 a, bf16x16 b, f32x8 c) {
  return __builtin_amdgcn_wmma_f32_16x16x32_bf16(false, a, false, b,
                                                 (short)0, c, false, false);
}

// LDS byte offset of a shared-memory pointer: generic (flat) shared addresses
// carry the LDS offset in addr[31:0] (aperture is in addr[63:32]).
__device__ __forceinline__ unsigned lds_off(const void* p) {
  return (unsigned)(unsigned long long)(uintptr_t)p;
}

// one async 512B chunk copy: 32 lanes x 16B, global -> LDS, tracked by ASYNCcnt
__device__ __forceinline__ void async_copy_b128(unsigned lds_addr,
                                                const unsigned short* gaddr) {
  unsigned long long ga = (unsigned long long)(uintptr_t)gaddr;
  asm volatile("global_load_async_to_lds_b128 %0, %1, off"
               :: "v"(lds_addr), "v"(ga) : "memory");
}

// ---------------- converters ----------------

__global__ void k_init_x(const float* __restrict__ x, float* __restrict__ xo,
                         unsigned short* __restrict__ xb, int n) {
  for (int i = blockIdx.x * blockDim.x + threadIdx.x; i < n;
       i += gridDim.x * blockDim.x) {
    float v = x[i];
    xo[i] = v;
    xb[i] = f2bf(v);
  }
}

// src (batch, rows, cols) fp32 -> dst (batch, cols, rows) bf16  (stores B^T)
__global__ void k_cvt_t(const float* __restrict__ src, unsigned short* __restrict__ dst,
                        int batch, int rows, int cols) {
  int n = batch * rows * cols;
  for (int i = blockIdx.x * blockDim.x + threadIdx.x; i < n;
       i += gridDim.x * blockDim.x) {
    int bt = i / (rows * cols);
    int rc = i - bt * rows * cols;
    int r = rc / cols, c = rc - r * cols;
    dst[bt * rows * cols + c * rows + r] = f2bf(src[i]);
  }
}

// ---------------- QKV projection ----------------
// grid (MROWS/16, NH, 3), block 64 (2 waves, wave -> dk half)
__global__ void k_qkv(const unsigned short* __restrict__ xb,
                      const unsigned short* __restrict__ wqt,
                      const unsigned short* __restrict__ wkt,
                      const unsigned short* __restrict__ wvt,
                      unsigned short* __restrict__ qb,
                      unsigned short* __restrict__ kb,
                      unsigned short* __restrict__ vbT) {
  const int m0   = blockIdx.x * 16;
  const int h    = blockIdx.y;
  const int z    = blockIdx.z;
  const int wave = threadIdx.x >> 5;
  const int lane = threadIdx.x & 31;
  const int dk0  = wave * 16;
  const unsigned short* wt = (z == 0) ? wqt : (z == 1) ? wkt : wvt;
  const unsigned short* wtile = wt + (h * DKV + dk0) * DMODEL;  // W^T (DK x D) per head

  f32x8 c = {};
  for (int k = 0; k < DMODEL; k += 32) {
    bf16x16 a = load_a_frag(xb + m0 * DMODEL + k, DMODEL);
    bf16x16 b = load_b_frag(wtile + k, DMODEL);
    c = wmma_bf16(a, b, c);
  }
  const int n = lane & 15;
  const int mloc = (lane >> 4) * 8;
#pragma unroll
  for (int r = 0; r < 8; ++r) {
    int m = m0 + mloc + r;
    int bidx = m >> 11;                   // / SEQ
    int s = m & (SEQ - 1);
    int bh = bidx * NH + h;
    unsigned short val = f2bf(c[r]);
    if (z == 2) {
      vbT[(bh * DKV + dk0 + n) * SEQ + s] = val;      // V stored transposed
    } else {
      unsigned short* dst = (z == 0) ? qb : kb;
      dst[(bh * SEQ + s) * DKV + dk0 + n] = val;
    }
  }
}

// ---------------- flash attention ----------------
// grid (SEQ/16, BDIM*NH), block 32 (one wave)
// K/V tiles are staged into LDS by the async copy engine, double-buffered:
// per 16-key step: 4 x global_load_async_to_lds_b128 (K tile 1KB + V tile 1KB)
__global__ void k_attn(const unsigned short* __restrict__ qb,
                       const unsigned short* __restrict__ kb,
                       const unsigned short* __restrict__ vbT,
                       const float* __restrict__ mask,
                       unsigned short* __restrict__ ob) {
  __shared__ __align__(16) unsigned short kbuf[2][16 * DKV];   // (key,dk) ld=32
  __shared__ __align__(16) unsigned short vbuf[2][DKV * 16];   // (dv,key) ld=16
  __shared__ __align__(16) unsigned short pl[16 * 16];
  const int s0 = blockIdx.x * 16;
  const int bh = blockIdx.y;
  const int b  = bh / NH, h = bh % NH;
  const int lane = threadIdx.x & 31;
  const int n    = lane & 15;
  const int half = lane >> 4;
  const unsigned short* qp = qb + (bh * SEQ + s0) * DKV;
  const unsigned short* kp = kb + bh * SEQ * DKV;
  const unsigned short* vp = vbT + bh * DKV * SEQ;

  // issue async staging of key-tile t into buffer bufi (4 async b128 ops)
  auto issue_tile = [&](int t, int bufi) {
    const unsigned kl = lds_off(&kbuf[bufi][0]);
    const unsigned vl = lds_off(&vbuf[bufi][0]);
    const unsigned short* kg = kp + t * DKV;
    // K tile: chunk c (16B) -> row c>>2, elem (c&3)*8 ; LDS linear
    int c0 = lane, c1 = lane + 32;
    async_copy_b128(kl + c0 * 16, kg + (c0 >> 2) * DKV + (c0 & 3) * 8);
    async_copy_b128(kl + c1 * 16, kg + (c1 >> 2) * DKV + (c1 & 3) * 8);
    // V tile: chunk c (16B) -> dv c>>1, key off (c&1)*8 ; LDS linear
    async_copy_b128(vl + c0 * 16, vp + (c0 >> 1) * SEQ + t + (c0 & 1) * 8);
    async_copy_b128(vl + c1 * 16, vp + (c1 >> 1) * SEQ + t + (c1 & 1) * 8);
  };

  bf16x16 qf = load_a_frag(qp, DKV);          // 16 queries x 32 dk (one WMMA K)
  f32x8 o0 = {}, o1 = {};
  float Mr[8], Lr[8];
#pragma unroll
  for (int r = 0; r < 8; ++r) { Mr[r] = -3.0e38f; Lr[r] = 0.f; }

  issue_tile(0, 0);
  for (int t = 0, it = 0; t < SEQ; t += 16, ++it) {
    const int cur = it & 1;
    const bool more = (t + 16) < SEQ;
    if (more) {
      issue_tile(t + 16, cur ^ 1);
      asm volatile("s_wait_asynccnt 0x4" ::: "memory");  // current tile resident
    } else {
      asm volatile("s_wait_asynccnt 0x0" ::: "memory");
    }
    __syncthreads();

    bf16x16 kf = load_a_frag(&kbuf[cur][0], DKV);  // from LDS
    f32x8 zc = {};
    f32x8 sc = wmma_bf16(qf, kf, zc);              // scores 16x16
    float mv = mask[b * SEQ + t + n];
#pragma unroll
    for (int r = 0; r < 8; ++r) {
      float sv = mv * sc[r] + (1.f - mv) * MASKV;
      float tmax = sv;                              // row-wise over 16 key lanes
      tmax = fmaxf(tmax, __shfl_xor(tmax, 1, 32));
      tmax = fmaxf(tmax, __shfl_xor(tmax, 2, 32));
      tmax = fmaxf(tmax, __shfl_xor(tmax, 4, 32));
      tmax = fmaxf(tmax, __shfl_xor(tmax, 8, 32));
      float newM = fmaxf(Mr[r], tmax);
      float corr = __expf(Mr[r] - newM);
      float p    = __expf(sv - newM);
      float psum = p;
      psum += __shfl_xor(psum, 1, 32);
      psum += __shfl_xor(psum, 2, 32);
      psum += __shfl_xor(psum, 4, 32);
      psum += __shfl_xor(psum, 8, 32);
      Lr[r] = Lr[r] * corr + psum;
      Mr[r] = newM;
      o0[r] *= corr;
      o1[r] *= corr;
      pl[(half * 8 + r) * 16 + n] = f2bf(p);        // C-layout -> LDS
    }
    __syncthreads();
    bf16x16 pf = load_a_frag(pl, 16);               // LDS -> A-layout (transpose)
    bf16x16 v0 = load_b_frag(&vbuf[cur][0], 16);    // V^T rows dv 0..15
    bf16x16 v1 = load_b_frag(&vbuf[cur][16 * 16], 16);
    o0 = wmma_bf16(pf, v0, o0);
    o1 = wmma_bf16(pf, v1, o1);
    __syncthreads();
  }
#pragma unroll
  for (int r = 0; r < 8; ++r) {
    float inv = 1.0f / Lr[r];
    int s = s0 + half * 8 + r;
    unsigned short* op = ob + (b * SEQ + s) * (NH * DKV) + h * DKV;
    op[n]      = f2bf(o0[r] * inv);
    op[16 + n] = f2bf(o1[r] * inv);
  }
}

// ---------------- FFN layer 1 (GEMM + bias + relu) ----------------
// grid (MROWS/16, DFFN/32), block 64
__global__ void k_ffn1(const unsigned short* __restrict__ xb,
                       const unsigned short* __restrict__ w1t,
                       const float* __restrict__ b1,
                       unsigned short* __restrict__ h1b) {
  const int m0   = blockIdx.x * 16;
  const int wave = threadIdx.x >> 5;
  const int lane = threadIdx.x & 31;
  const int n0   = blockIdx.y * 32 + wave * 16;
  f32x8 c = {};
  for (int k = 0; k < DMODEL; k += 32) {
    __builtin_prefetch(xb + m0 * DMODEL + k + 64, 0, 1);
    bf16x16 a = load_a_frag(xb + m0 * DMODEL + k, DMODEL);
    bf16x16 b = load_b_frag(w1t + n0 * DMODEL + k, DMODEL);
    c = wmma_bf16(a, b, c);
  }
  const int n = n0 + (lane & 15);
  const int mloc = (lane >> 4) * 8;
  const float bias = b1[n];
#pragma unroll
  for (int r = 0; r < 8; ++r) {
    float v = c[r] + bias;
    v = v > 0.f ? v : 0.f;
    h1b[(m0 + mloc + r) * DFFN + n] = f2bf(v);
  }
}

// ------ GEMM (+bias) + residual + layernorm, N = DMODEL fixed ------
// grid (MROWS/16), block 128 (4 waves x 4 n-tiles)
__global__ void k_gemm_resid_ln(const unsigned short* __restrict__ A,
                                const unsigned short* __restrict__ Bt,
                                const float* __restrict__ bias,
                                float* __restrict__ x,
                                unsigned short* __restrict__ xb,
                                int K,
                                const float* __restrict__ gamma,
                                const float* __restrict__ beta,
                                int gi) {
  __shared__ float ls[16 * DMODEL];
  const int m0   = blockIdx.x * 16;
  const int wave = threadIdx.x >> 5;
  const int lane = threadIdx.x & 31;
  f32x8 c[4] = {{}, {}, {}, {}};
  for (int k = 0; k < K; k += 32) {
    __builtin_prefetch(A + m0 * K + k + 64, 0, 1);
    bf16x16 a = load_a_frag(A + m0 * K + k, K);
#pragma unroll
    for (int j = 0; j < 4; ++j) {
      bf16x16 b = load_b_frag(Bt + (wave * 64 + j * 16) * K + k, K);
      c[j] = wmma_bf16(a, b, c[j]);
    }
  }
  const int nn   = lane & 15;
  const int mloc = (lane >> 4) * 8;
#pragma unroll
  for (int j = 0; j < 4; ++j) {
    int n = wave * 64 + j * 16 + nn;
    float bv = bias ? bias[n] : 0.f;
#pragma unroll
    for (int r = 0; r < 8; ++r) {
      int m = m0 + mloc + r;
      ls[(mloc + r) * DMODEL + n] = c[j][r] + bv + x[m * DMODEL + n];
    }
  }
  __syncthreads();
  // layernorm: 16 rows, 8 threads/row, 32 cols/thread
  const int tid  = threadIdx.x;
  const int row  = tid >> 3;
  const int part = tid & 7;
  const float* rp = ls + row * DMODEL;
  float s = 0.f;
#pragma unroll
  for (int cidx = 0; cidx < 32; ++cidx) s += rp[part * 32 + cidx];
  s += __shfl_xor(s, 1, 32);
  s += __shfl_xor(s, 2, 32);
  s += __shfl_xor(s, 4, 32);
  const float mean = s * (1.0f / DMODEL);
  float vs = 0.f;
#pragma unroll
  for (int cidx = 0; cidx < 32; ++cidx) {
    float d = rp[part * 32 + cidx] - mean;
    vs += d * d;
  }
  vs += __shfl_xor(vs, 1, 32);
  vs += __shfl_xor(vs, 2, 32);
  vs += __shfl_xor(vs, 4, 32);
  const float rstd = rsqrtf(vs * (1.0f / DMODEL) + LN_EPS);
  const float g = gamma[gi], be = beta[gi];
  const int m = m0 + row;
#pragma unroll
  for (int cidx = 0; cidx < 32; ++cidx) {
    int col = part * 32 + cidx;
    float y = (rp[col] - mean) * rstd * g + be;
    x[m * DMODEL + col]  = y;
    xb[m * DMODEL + col] = f2bf(y);
  }
}

// ---------------- host launcher ----------------

extern "C" void kernel_launch(void* const* d_in, const int* in_sizes, int n_in,
                              void* d_out, int out_size, void* d_ws, size_t ws_size,
                              hipStream_t stream) {
  const float* x     = (const float*)d_in[0];
  const float* mask  = (const float*)d_in[1];
  const float* Wq    = (const float*)d_in[2];
  const float* Wk    = (const float*)d_in[3];
  const float* Wv    = (const float*)d_in[4];
  const float* Wo    = (const float*)d_in[5];
  const float* W1    = (const float*)d_in[6];
  const float* b1    = (const float*)d_in[7];
  const float* W2    = (const float*)d_in[8];
  const float* b2    = (const float*)d_in[9];
  const float* gamma = (const float*)d_in[10];
  const float* beta  = (const float*)d_in[11];
  float* xout = (float*)d_out;

  char* w = (char*)d_ws;
  auto alloc = [&](size_t bytes) -> char* {
    char* p = w;
    w += (bytes + 255) & ~(size_t)255;
    return p;
  };
  unsigned short* xb  = (unsigned short*)alloc((size_t)MROWS * DMODEL * 2);
  unsigned short* qb  = (unsigned short*)alloc((size_t)BDIM * NH * SEQ * DKV * 2);
  unsigned short* kb  = (unsigned short*)alloc((size_t)BDIM * NH * SEQ * DKV * 2);
  unsigned short* vbT = (unsigned short*)alloc((size_t)BDIM * NH * SEQ * DKV * 2);
  unsigned short* ob  = (unsigned short*)alloc((size_t)MROWS * (NH * DKV) * 2);
  unsigned short* h1b = (unsigned short*)alloc((size_t)MROWS * DFFN * 2);
  unsigned short* wqt = (unsigned short*)alloc((size_t)NH * DMODEL * DKV * 2);
  unsigned short* wkt = (unsigned short*)alloc((size_t)NH * DMODEL * DKV * 2);
  unsigned short* wvt = (unsigned short*)alloc((size_t)NH * DMODEL * DKV * 2);
  unsigned short* wot = (unsigned short*)alloc((size_t)(NH * DKV) * DMODEL * 2);
  unsigned short* w1t = (unsigned short*)alloc((size_t)DMODEL * DFFN * 2);
  unsigned short* w2t = (unsigned short*)alloc((size_t)DFFN * DMODEL * 2);
  (void)ws_size; (void)in_sizes; (void)n_in; (void)out_size;

  k_init_x<<<1024, 256, 0, stream>>>(x, xout, xb, MROWS * DMODEL);

  const int wqkv_sz = NH * DMODEL * DKV;       // 65536 per layer
  const int wo_sz   = (NH * DKV) * DMODEL;     // 65536 per layer
  const int w12_sz  = DMODEL * DFFN;           // 131072 per layer

  for (int i = 0; i < NLAYER; ++i) {
    k_cvt_t<<<256, 256, 0, stream>>>(Wq + (size_t)i * wqkv_sz, wqt, NH, DMODEL, DKV);
    k_cvt_t<<<256, 256, 0, stream>>>(Wk + (size_t)i * wqkv_sz, wkt, NH, DMODEL, DKV);
    k_cvt_t<<<256, 256, 0, stream>>>(Wv + (size_t)i * wqkv_sz, wvt, NH, DMODEL, DKV);
    k_cvt_t<<<256, 256, 0, stream>>>(Wo + (size_t)i * wo_sz,   wot, 1, NH * DKV, DMODEL);
    k_cvt_t<<<256, 256, 0, stream>>>(W1 + (size_t)i * w12_sz,  w1t, 1, DMODEL, DFFN);
    k_cvt_t<<<256, 256, 0, stream>>>(W2 + (size_t)i * w12_sz,  w2t, 1, DFFN, DMODEL);

    k_qkv<<<dim3(MROWS / 16, NH, 3), 64, 0, stream>>>(xb, wqt, wkt, wvt, qb, kb, vbT);
    k_attn<<<dim3(SEQ / 16, BDIM * NH), 32, 0, stream>>>(qb, kb, vbT, mask, ob);
    k_gemm_resid_ln<<<MROWS / 16, 128, 0, stream>>>(ob, wot, nullptr, xout, xb,
                                                    DMODEL, gamma, beta, 2 * i);
    k_ffn1<<<dim3(MROWS / 16, DFFN / 32), 64, 0, stream>>>(xb, w1t, b1 + (size_t)i * DFFN, h1b);
    k_gemm_resid_ln<<<MROWS / 16, 128, 0, stream>>>(h1b, w2t, b2 + (size_t)i * DMODEL,
                                                    xout, xb, DFFN, gamma, beta, 2 * i + 1);
  }
}
